// WindowAttention_44590350467397
// MI455X (gfx1250) — matrix operand order repaired and checked by reading, not compiled, not run on hardware
//
#include <hip/hip_runtime.h>

typedef __attribute__((ext_vector_type(16))) __bf16 v16bf;
typedef __attribute__((ext_vector_type(8)))  float  v8f;

#define BN   128
#define NSEQ 343
#define NPAD 352
#define CDIM 512
#define NH   16
#define HD   32
#define CO   1536
#define ROWS (BN * NSEQ)      // 43904, divisible by 16
#define MT   (ROWS / 16)      // 2744
#define QT   22               // ceil(343/16), 22*16 = 352 = NPAD
#define QSCALE 0.17677669529663687f  // 32^-0.5

static __device__ __forceinline__ v8f wmma_bf16(v16bf a, v16bf b, v8f c) {
  return __builtin_amdgcn_wmma_f32_16x16x32_bf16(false, a, false, b, (short)0, c, false, false);
}

// ---------------------------------------------------------------- fp32 -> bf16
__global__ void cvt_f32_bf16(const float* __restrict__ src,
                             unsigned short* __restrict__ dstRaw, int n) {
  int i = blockIdx.x * blockDim.x + threadIdx.x;
  __bf16* dst = reinterpret_cast<__bf16*>(dstRaw);
  if (i < n) dst[i] = (__bf16)src[i];
}

// ------------------------------------------------- QKV GEMM + head scatter
// One wave computes a 16x64 tile of qkv = x @ W^T + b; scatters into
// q/k/v buffers laid out [B, H, NPAD, HD] bf16 (q pre-scaled).
__global__ void __launch_bounds__(32) qkv_gemm(
    const float* __restrict__ x, const unsigned short* __restrict__ wRaw,
    const float* __restrict__ bias,
    unsigned short* __restrict__ qRaw, unsigned short* __restrict__ kRaw,
    unsigned short* __restrict__ vRaw) {
  const __bf16* w = reinterpret_cast<const __bf16*>(wRaw);
  __bf16* qb = reinterpret_cast<__bf16*>(qRaw);
  __bf16* kb = reinterpret_cast<__bf16*>(kRaw);
  __bf16* vb = reinterpret_cast<__bf16*>(vRaw);

  int blk = blockIdx.x;
  int mt = blk / (CO / 64);
  int ct = blk % (CO / 64);
  int lane = threadIdx.x;
  int lm = lane & 15, lh = lane >> 4;
  int row = mt * 16 + lm;                       // always < ROWS

  v8f acc[4] = {};
  for (int ks = 0; ks < CDIM / 32; ++ks) {
    const float* xr = x + (size_t)row * CDIM + ks * 32;
    v16bf a;
#pragma unroll
    for (int i = 0; i < 16; ++i) {
      a[i]     = (__bf16)xr[lh * 16 + i];       // K = lh*16 + i
    }
#pragma unroll
    for (int nt = 0; nt < 4; ++nt) {
      int n = ct * 64 + nt * 16 + lm;
      const __bf16* wr = w + (size_t)n * CDIM + ks * 32 + lh * 16;
      v16bf bf;
#pragma unroll
      for (int i = 0; i < 16; ++i) bf[i] = wr[i];   // B[k][n] = W[n][k]
      acc[nt] = wmma_bf16(a, bf, acc[nt]);
    }
  }
#pragma unroll
  for (int nt = 0; nt < 4; ++nt) {
    int col = ct * 64 + nt * 16 + lm;
    int which = col / CDIM;
    int rem = col - which * CDIM;
    int h = rem >> 5, d = rem & 31;
    float bcol = bias[col];
    __bf16* dst = (which == 0) ? qb : ((which == 1) ? kb : vb);
    float sc = (which == 0) ? QSCALE : 1.0f;
#pragma unroll
    for (int r = 0; r < 8; ++r) {
      int rm = mt * 16 + r + 8 * lh;
      int b = rm / NSEQ;
      int nn = rm - b * NSEQ;
      float val = (acc[nt][r] + bcol) * sc;
      dst[((size_t)(b * NH + h) * NPAD + nn) * HD + d] = (__bf16)val;
    }
  }
}

// ----------------------------------------------------------- attention core
// One wave handles one (b, h, 16-query tile): S -> bias -> softmax -> P@V.
__global__ void __launch_bounds__(32) attn_kernel(
    const unsigned short* __restrict__ qRaw, const unsigned short* __restrict__ kRaw,
    const unsigned short* __restrict__ vRaw,
    const float* __restrict__ table, const int* __restrict__ rpi,
    unsigned short* __restrict__ aoRaw) {
  __shared__ float  sS[16 * NPAD];   // scores, fp32
  __shared__ __bf16 sP[16 * NPAD];   // softmax probs, bf16

  const __bf16* q = reinterpret_cast<const __bf16*>(qRaw);
  const __bf16* k = reinterpret_cast<const __bf16*>(kRaw);
  const __bf16* v = reinterpret_cast<const __bf16*>(vRaw);
  __bf16* ao = reinterpret_cast<__bf16*>(aoRaw);

  int blk = blockIdx.x;
  int qt = blk % QT;
  int bh = blk / QT;
  int h = bh & (NH - 1);
  int b = bh / NH;
  int lane = threadIdx.x;
  int lm = lane & 15, lh = lane >> 4;

  const __bf16* qbase = q + (size_t)bh * NPAD * HD;
  const __bf16* kbase = k + (size_t)bh * NPAD * HD;
  const __bf16* vbase = v + (size_t)bh * NPAD * HD;

  // Q A-fragment (hd = 32 -> single K step)
  v16bf aq;
  const __bf16* qr = qbase + (qt * 16 + lm) * HD;
#pragma unroll
  for (int i = 0; i < 16; ++i) {
    aq[i]     = qr[lh * 16 + i];
  }

  // S = q @ k^T, plus relative-position bias, into LDS
  for (int jt = 0; jt < QT; ++jt) {
    v16bf bk;
    const __bf16* kr = kbase + (jt * 16 + lm) * HD + lh * 16;
#pragma unroll
    for (int i = 0; i < 16; ++i) bk[i] = kr[i];   // B[kd][n] = K[jt*16+n][kd]
    v8f c = {};
    c = wmma_bf16(aq, bk, c);
    int kj = jt * 16 + lm;
#pragma unroll
    for (int r = 0; r < 8; ++r) {
      int m = r + 8 * lh;
      int qi = qt * 16 + m;
      float s;
      if (kj < NSEQ) {
        s = c[r];
        if (qi < NSEQ) s += table[rpi[qi * NSEQ + kj] * NH + h];
      } else {
        s = -1e30f;                                // mask padded keys
      }
      sS[m * NPAD + jt * 16 + lm] = s;
    }
  }
  __syncthreads();

  // row softmax (lanes 0..15, one row each)
  if (lane < 16) {
    float* rowp = sS + lane * NPAD;
    float mx = -1e30f;
    for (int j = 0; j < NPAD; ++j) mx = fmaxf(mx, rowp[j]);
    float sum = 0.0f;
    for (int j = 0; j < NPAD; ++j) { float e = __expf(rowp[j] - mx); sum += e; rowp[j] = e; }
    float inv = 1.0f / sum;
    __bf16* prow = sP + lane * NPAD;
    for (int j = 0; j < NPAD; ++j) prow[j] = (__bf16)(rowp[j] * inv);
  }
  __syncthreads();

  // out = P @ V   (16 x 32, K = 352 -> 11 WMMA steps x 2 column tiles)
  v8f o0 = {}, o1 = {};
  for (int ks = 0; ks < NPAD / 32; ++ks) {
    v16bf ap;
    const __bf16* pr = sP + lm * NPAD + ks * 32;
#pragma unroll
    for (int i = 0; i < 16; ++i) {
      ap[i]     = pr[lh * 16 + i];
    }
    const __bf16* vr = vbase + (size_t)(ks * 32 + lh * 16) * HD;
    v16bf bv0, bv1;
#pragma unroll
    for (int i = 0; i < 16; ++i) {
      bv0[i] = vr[i * HD + lm];          // B[kd][n] = V[kglob][n]
      bv1[i] = vr[i * HD + 16 + lm];     // second 16 cols of hd
    }
    o0 = wmma_bf16(ap, bv0, o0);
    o1 = wmma_bf16(ap, bv1, o1);
  }

  // store to flat [ROWS, CDIM] bf16, row = b*343 + qi, col = h*32 + d
#pragma unroll
  for (int r = 0; r < 8; ++r) {
    int qi = qt * 16 + r + 8 * lh;
    if (qi < NSEQ) {
      size_t off = (size_t)(b * NSEQ + qi) * CDIM + h * HD;
      ao[off + lm]      = (__bf16)o0[r];
      ao[off + 16 + lm] = (__bf16)o1[r];
    }
  }
}

// ----------------------------------------------------------- output projection
__global__ void __launch_bounds__(32) proj_gemm(
    const unsigned short* __restrict__ aoRaw, const unsigned short* __restrict__ wRaw,
    const float* __restrict__ bias, float* __restrict__ out) {
  const __bf16* ao = reinterpret_cast<const __bf16*>(aoRaw);
  const __bf16* w  = reinterpret_cast<const __bf16*>(wRaw);

  int blk = blockIdx.x;
  int mt = blk / (CDIM / 64);
  int ct = blk % (CDIM / 64);
  int lane = threadIdx.x;
  int lm = lane & 15, lh = lane >> 4;
  int row = mt * 16 + lm;

  v8f acc[4] = {};
  for (int ks = 0; ks < CDIM / 32; ++ks) {
    const __bf16* ar = ao + (size_t)row * CDIM + ks * 32;
    v16bf a;
#pragma unroll
    for (int i = 0; i < 16; ++i) {
      a[i]     = ar[lh * 16 + i];
    }
#pragma unroll
    for (int nt = 0; nt < 4; ++nt) {
      int n = ct * 64 + nt * 16 + lm;
      const __bf16* wr = w + (size_t)n * CDIM + ks * 32 + lh * 16;
      v16bf bf;
#pragma unroll
      for (int i = 0; i < 16; ++i) bf[i] = wr[i];
      acc[nt] = wmma_bf16(a, bf, acc[nt]);
    }
  }
#pragma unroll
  for (int nt = 0; nt < 4; ++nt) {
    int col = ct * 64 + nt * 16 + lm;
    float bcol = bias[col];
#pragma unroll
    for (int r = 0; r < 8; ++r) {
      int rm = mt * 16 + r + 8 * lh;
      out[(size_t)rm * CDIM + col] = acc[nt][r] + bcol;
    }
  }
}

// --------------------------------------------------------------------- launch
extern "C" void kernel_launch(void* const* d_in, const int* in_sizes, int n_in,
                              void* d_out, int out_size, void* d_ws, size_t ws_size,
                              hipStream_t stream) {
  const float* x        = (const float*)d_in[0];
  // d_in[1] q_global is unused by the reference
  const float* qkv_w    = (const float*)d_in[2];
  const float* qkv_bias = (const float*)d_in[3];
  const float* table    = (const float*)d_in[4];
  const float* proj_w   = (const float*)d_in[5];
  const float* proj_bias= (const float*)d_in[6];
  const int*   rpi      = (const int*)d_in[7];
  float* out = (float*)d_out;

  // workspace carve-out (~186 MB)
  char* p = (char*)d_ws;
  unsigned short* wqkvb  = (unsigned short*)p; p += (size_t)CO   * CDIM * 2;
  unsigned short* wprojb = (unsigned short*)p; p += (size_t)CDIM * CDIM * 2;
  size_t headBytes = (size_t)BN * NH * NPAD * HD * 2;
  unsigned short* qbuf = (unsigned short*)p; p += headBytes;
  unsigned short* kbuf = (unsigned short*)p; p += headBytes;
  unsigned short* vbuf = (unsigned short*)p; p += headBytes;
  unsigned short* aobuf = (unsigned short*)p; p += (size_t)ROWS * CDIM * 2;

  // zero q/k/v padding rows (NPAD > NSEQ) so WMMA tiles are clean
  hipMemsetAsync(qbuf, 0, 3 * headBytes, stream);

  cvt_f32_bf16<<<(CO * CDIM + 255) / 256, 256, 0, stream>>>(qkv_w, wqkvb, CO * CDIM);
  cvt_f32_bf16<<<(CDIM * CDIM + 255) / 256, 256, 0, stream>>>(proj_w, wprojb, CDIM * CDIM);

  qkv_gemm<<<MT * (CO / 64), 32, 0, stream>>>(x, wqkvb, qkv_bias, qbuf, kbuf, vbuf);
  attn_kernel<<<BN * NH * QT, 32, 0, stream>>>(qbuf, kbuf, vbuf, table, rpi, aobuf);
  proj_gemm<<<MT * (CDIM / 64), 32, 0, stream>>>(aobuf, wprojb, proj_bias, out);
}
